// FewShotABSALearner_46935402610855
// MI455X (gfx1250) — matrix operand order repaired and checked by reading, not compiled, not run on hardware
//
#include <hip/hip_runtime.h>
#include <hip/hip_bf16.h>

typedef __bf16 bf16_t;
typedef __attribute__((ext_vector_type(16))) __bf16   v16bf;
typedef __attribute__((ext_vector_type(8)))  float    v8f;
typedef __attribute__((ext_vector_type(4)))  unsigned uint4e;
typedef __attribute__((ext_vector_type(4)))  int      v4i;

constexpr int HD   = 1024;
constexpr int NSEN = 32;          // NS
constexpr int NQRY = 64;          // NQ
constexpr int SEQ  = 256;         // S
constexpr int NTOK = NQRY * SEQ;  // 16384 query tokens
constexpr int NSUP = NSEN * SEQ;  // 8192 support tokens

// ---- async global->LDS path (gfx1250), guarded so fallback always compiles ----
#if __has_builtin(__builtin_amdgcn_global_load_async_to_lds_b128) && \
    __has_builtin(__builtin_amdgcn_s_wait_asynccnt)
#define HAVE_ASYNC_LDS 1
#else
#define HAVE_ASYNC_LDS 0
#endif

#if HAVE_ASYNC_LDS
#define AS1 __attribute__((address_space(1)))
#define AS3 __attribute__((address_space(3)))
__device__ __forceinline__ void async_copy_b128(const void* gsrc, void* ldst) {
  __builtin_amdgcn_global_load_async_to_lds_b128((AS1 v4i*)gsrc, (AS3 v4i*)ldst, 0, 0);
}
#endif

// ---------------- f32 -> bf16 conversion (grid-stride), row-major ----------------
__global__ void k_f2bf(const float* __restrict__ in, bf16_t* __restrict__ out, long n) {
  long i = (long)blockIdx.x * blockDim.x + threadIdx.x;
  long stride = (long)gridDim.x * blockDim.x;
  for (; i < n; i += stride) out[i] = (bf16_t)in[i];
}

// ------- f32 weights -> fragment-native packed bf16: PB[j][n] = (B[2j][n], B[2j+1][n]) -------
__global__ void k_f2bf_pack(const float* __restrict__ in, unsigned* __restrict__ out,
                            int K, int N) {
  int total = (K / 2) * N;
  int idx = blockIdx.x * blockDim.x + threadIdx.x;
  if (idx >= total) return;
  int j = idx / N, n = idx - j * N;
  union { bf16_t b[2]; unsigned u; } p;
  p.b[0] = (bf16_t)in[(size_t)(2 * j) * N + n];
  p.b[1] = (bf16_t)in[(size_t)(2 * j + 1) * N + n];
  out[idx] = p.u;
}

// ---------------- WMMA bf16 GEMM: C = act(A[M,K] @ B[K,N] + bias) ----------------
// A: row-major bf16.  B: pre-packed PB[K/2][N] dwords (bf16 k-pair per dword).
constexpr int BM = 128, BN = 64, BK = 32;
constexpr int APAD = 8;        // As row stride 40 elems = 80B (16B multiple, conflict-free)
constexpr int BKP  = 20;       // Bs row stride 20 dwords = 80B (16B multiple, conflict-free)
#if HAVE_ASYNC_LDS
constexpr int NBUF = 2;
#else
constexpr int NBUF = 1;
#endif

union FragAB { v16bf v; uint4e q[2]; };

template<int ACT, int OUTBF>
__launch_bounds__(256)
__global__ void k_gemm(const bf16_t* __restrict__ A, const unsigned* __restrict__ Bpk,
                       const float* __restrict__ bias, void* __restrict__ Cout,
                       int M, int N, int K) {
  __shared__ __align__(16) bf16_t   As[NBUF][BM][BK + APAD];
  __shared__ __align__(16) unsigned Bs[NBUF][BN][BKP];     // transposed: [col][kpair]

  const int tid  = threadIdx.x;
  const int lane = tid & 31;
  const int wave = tid >> 5;     // 8 waves
  const int wm   = wave >> 1;    // 0..3  (M direction)
  const int wn   = wave & 1;     // 0..1  (N direction)
  const int m0   = blockIdx.y * BM;
  const int n0   = blockIdx.x * BN;

  // staging coordinates
  const int ar0 = tid >> 2;             // A rows 0..63 (pass 0), +64 (pass 1)
  const int ac  = (tid & 3) << 3;       // A col 0,8,16,24
  const int bjr = tid >> 4;             // B kpair row 0..15
  const int bcc = (tid & 15) << 2;      // B col 0..60 step 4

  // fragment coordinates
  const int arowl  = lane & 15;
  const int kha    = (lane < 16) ? 0 : 8;   // A K-half (elements)
  const int kp0    = (lane < 16) ? 0 : 4;   // B K-half (kpairs)
  const int coll   = lane & 15;
  const int rowoff = 8 * (lane >> 4);

  v8f acc[2][2] = {};

  const bf16_t*   Arow0 = A + (size_t)(m0 + ar0) * K + ac;
  const bf16_t*   Arow1 = A + (size_t)(m0 + ar0 + 64) * K + ac;
  const unsigned* Bsrc  = Bpk + (size_t)bjr * N + n0 + bcc;

  uint4e bReg;

  auto store_b = [&](int b) {
    Bs[b][bcc + 0][bjr] = bReg.x;
    Bs[b][bcc + 1][bjr] = bReg.y;
    Bs[b][bcc + 2][bjr] = bReg.z;
    Bs[b][bcc + 3][bjr] = bReg.w;
  };
  auto compute_tile = [&](int b) {
    FragAB af[2], bfr[2];
#pragma unroll
    for (int mt = 0; mt < 2; ++mt) {
      int arow = wm * 32 + mt * 16 + arowl;
      af[mt].q[0] = *(const uint4e*)&As[b][arow][kha];
      af[mt].q[1] = *(const uint4e*)&As[b][arow][16 + kha];
    }
#pragma unroll
    for (int nt = 0; nt < 2; ++nt) {
      int bcol = wn * 32 + nt * 16 + coll;
      bfr[nt].q[0] = *(const uint4e*)&Bs[b][bcol][kp0];
      bfr[nt].q[1] = *(const uint4e*)&Bs[b][bcol][kp0 + 8];
    }
#pragma unroll
    for (int mt = 0; mt < 2; ++mt)
#pragma unroll
      for (int nt = 0; nt < 2; ++nt)
        acc[mt][nt] = __builtin_amdgcn_wmma_f32_16x16x32_bf16(
            false, af[mt].v, false, bfr[nt].v, (short)0, acc[mt][nt], false, false);
  };

#if HAVE_ASYNC_LDS
  // ---- async double-buffered pipeline, last iteration peeled ----
  async_copy_b128(Arow0, &As[0][ar0][ac]);
  async_copy_b128(Arow1, &As[0][ar0 + 64][ac]);
  bReg = *(const uint4e*)Bsrc;

  int buf = 0;
  for (int k0 = 0; k0 < K - BK; k0 += BK) {
    const int nbuf = buf ^ 1;
    const int kn = k0 + BK;
    async_copy_b128(Arow0 + kn, &As[nbuf][ar0][ac]);
    async_copy_b128(Arow1 + kn, &As[nbuf][ar0 + 64][ac]);
    store_b(buf);
    bReg = *(const uint4e*)(Bsrc + (size_t)(kn / 2) * N);
    __builtin_amdgcn_s_wait_asynccnt(2);
    __syncthreads();
    compute_tile(buf);
    __syncthreads();
    buf = nbuf;
  }
  // final tile: nothing left to prefetch
  store_b(buf);
  __builtin_amdgcn_s_wait_asynccnt(0);
  __syncthreads();
  compute_tile(buf);
#else
  // ---- register-prefetch double buffering (fallback) ----
  uint4e aReg0 = *(const uint4e*)Arow0;
  uint4e aReg1 = *(const uint4e*)Arow1;
  bReg = *(const uint4e*)Bsrc;

  for (int k0 = 0; k0 < K; k0 += BK) {
    *(uint4e*)&As[0][ar0][ac]      = aReg0;
    *(uint4e*)&As[0][ar0 + 64][ac] = aReg1;
    store_b(0);
    __syncthreads();
    if (k0 + BK < K) {
      int kn = k0 + BK;
      aReg0 = *(const uint4e*)(Arow0 + kn);
      aReg1 = *(const uint4e*)(Arow1 + kn);
      bReg  = *(const uint4e*)(Bsrc + (size_t)(kn / 2) * N);
    }
    compute_tile(0);
    __syncthreads();
  }
#endif

  // ---- epilogue: bias + optional relu, one output format ----
#pragma unroll
  for (int mt = 0; mt < 2; ++mt)
#pragma unroll
    for (int nt = 0; nt < 2; ++nt) {
      int row0 = m0 + wm * 32 + mt * 16 + rowoff;
      int col  = n0 + wn * 32 + nt * 16 + coll;
      float bc = bias[col];
      size_t base = (size_t)row0 * N + col;
#pragma unroll
      for (int r = 0; r < 8; ++r) {
        float v = acc[mt][nt][r] + bc;
        if (ACT) v = v > 0.f ? v : 0.f;
        if (OUTBF) ((bf16_t*)Cout)[base + (size_t)r * N] = (bf16_t)v;
        else       ((float*) Cout)[base + (size_t)r * N] = v;
      }
    }
}

// ---------------- LayerNorm: one 256-thread block per row of HD ----------------
template<int OUTBF>
__global__ void k_ln(const float* __restrict__ in, const float* __restrict__ g,
                     const float* __restrict__ be, void* __restrict__ outp,
                     int ldout, int coloff) {
  const int row = blockIdx.x;
  const int t   = threadIdx.x;           // 256 threads, 4 elems each
  const float4* in4 = (const float4*)(in + (size_t)row * HD);
  float4 x = in4[t];

  __shared__ float red[256];
  red[t] = x.x + x.y + x.z + x.w;
  __syncthreads();
  for (int o = 128; o > 0; o >>= 1) { if (t < o) red[t] += red[t + o]; __syncthreads(); }
  float mu = red[0] * (1.0f / HD);
  __syncthreads();

  float d0 = x.x - mu, d1 = x.y - mu, d2 = x.z - mu, d3 = x.w - mu;
  red[t] = d0 * d0 + d1 * d1 + d2 * d2 + d3 * d3;
  __syncthreads();
  for (int o = 128; o > 0; o >>= 1) { if (t < o) red[t] += red[t + o]; __syncthreads(); }
  float rstd = rsqrtf(red[0] * (1.0f / HD) + 1e-5f);

  float dv[4] = {d0, d1, d2, d3};
#pragma unroll
  for (int j = 0; j < 4; ++j) {
    int h = t * 4 + j;
    float v = g[h] * dv[j] * rstd + be[h];
    if (OUTBF) ((bf16_t*)outp)[(size_t)row * ldout + coloff + h] = (bf16_t)v;
    else       ((float*) outp)[(size_t)row * ldout + coloff + h] = v;
  }
}

// ---------------- prototypes: masked mean over S per sentence ----------------
__global__ void k_proto(const float* __restrict__ awS, const int* __restrict__ labels,
                        float* __restrict__ protos) {
  int ns = blockIdx.x;                       // 0..31
  int h  = blockIdx.y * 256 + threadIdx.x;   // 0..1023
  float acc = 0.f; int cnt = 0;
  for (int s = 0; s < SEQ; ++s) {
    int lab = labels[ns * SEQ + s];
    if (lab > 0) { ++cnt; acc += awS[((size_t)ns * SEQ + s) * HD + h]; }
  }
  float c = (float)cnt; if (c < 1.f) c = 1.f;
  protos[ns * HD + h] = acc / c;
}

__global__ void k_pnorm(const float* __restrict__ protos, float* __restrict__ pn) {
  int p = blockIdx.x, t = threadIdx.x;
  const float4* pr = (const float4*)(protos + (size_t)p * HD);
  float4 x = pr[t];
  __shared__ float red[256];
  red[t] = x.x * x.x + x.y * x.y + x.z * x.z + x.w * x.w;
  __syncthreads();
  for (int o = 128; o > 0; o >>= 1) { if (t < o) red[t] += red[t + o]; __syncthreads(); }
  float rn = 1.f / fmaxf(sqrtf(red[0]), 1e-8f);
  float4* o4 = (float4*)(pn + (size_t)p * HD);
  float4 y; y.x = x.x * rn; y.y = x.y * rn; y.z = x.z * rn; y.w = x.w * rn;
  o4[t] = y;
}

// ------ fused adaptation attention: x' = adapted + softmax(cos-sims)@protos, bf16 out ------
__global__ void k_attn(const float* __restrict__ adapted, const float* __restrict__ protos,
                       const float* __restrict__ pn, bf16_t* __restrict__ xp) {
  const int tok  = blockIdx.x;
  const int t    = threadIdx.x;            // 256 threads, 4 H-elems each
  const int lane = t & 31;
  const int wave = t >> 5;

  const float4* a4  = (const float4*)(adapted + (size_t)tok * HD);
  const float4* pn4 = (const float4*)pn;
  float4 x = a4[t];

  float part[NSEN];
  float ss = x.x * x.x + x.y * x.y + x.z * x.z + x.w * x.w;
#pragma unroll
  for (int p = 0; p < NSEN; ++p) {
    float4 v = pn4[p * (HD / 4) + t];
    part[p] = x.x * v.x + x.y * v.y + x.z * v.z + x.w * v.w;
  }
#pragma unroll
  for (int p = 0; p < NSEN; ++p)
#pragma unroll
    for (int m = 16; m > 0; m >>= 1) part[p] += __shfl_xor(part[p], m, 32);
#pragma unroll
  for (int m = 16; m > 0; m >>= 1) ss += __shfl_xor(ss, m, 32);

  __shared__ float wred[NSEN + 1][8];
  if (lane == 0) {
#pragma unroll
    for (int p = 0; p < NSEN; ++p) wred[p][wave] = part[p];
    wred[NSEN][wave] = ss;
  }
  __syncthreads();

  __shared__ float wts[NSEN];
  if (t == 0) {
    float sims[NSEN];
    float sstot = 0.f;
    for (int w = 0; w < 8; ++w) sstot += wred[NSEN][w];
    float rn = 1.f / fmaxf(sqrtf(sstot), 1e-8f);
    float mx = -3.0e38f;
    for (int p = 0; p < NSEN; ++p) {
      float s = 0.f;
      for (int w = 0; w < 8; ++w) s += wred[p][w];
      s *= rn; sims[p] = s; mx = fmaxf(mx, s);
    }
    float den = 0.f;
    for (int p = 0; p < NSEN; ++p) { float e = __expf(sims[p] - mx); sims[p] = e; den += e; }
    float rden = 1.f / den;
    for (int p = 0; p < NSEN; ++p) wts[p] = sims[p] * rden;
  }
  __syncthreads();

  const float4* pr4 = (const float4*)protos;
  float ix = 0.f, iy = 0.f, iz = 0.f, iw = 0.f;
#pragma unroll
  for (int p = 0; p < NSEN; ++p) {
    float w = wts[p];
    float4 v = pr4[p * (HD / 4) + t];
    ix += w * v.x; iy += w * v.y; iz += w * v.z; iw += w * v.w;
  }
  bf16_t* o = xp + (size_t)tok * HD + t * 4;
  o[0] = (bf16_t)(x.x + ix);
  o[1] = (bf16_t)(x.y + iy);
  o[2] = (bf16_t)(x.z + iz);
  o[3] = (bf16_t)(x.w + iw);
}

// ---------------- final tiny projection: [T,64] @ [64,3] + b ----------------
__global__ void k_final(const float* __restrict__ hf, const float* __restrict__ W3,
                        const float* __restrict__ b3, float* __restrict__ out, int T) {
  int idx = blockIdx.x * blockDim.x + threadIdx.x;
  if (idx >= T * 3) return;
  int t = idx / 3, c = idx % 3;
  float acc = b3[c];
  for (int k = 0; k < 64; ++k) acc += hf[(size_t)t * 64 + k] * W3[k * 3 + c];
  out[idx] = acc;
}

// ============================== host launcher ==============================
extern "C" void kernel_launch(void* const* d_in, const int* in_sizes, int n_in,
                              void* d_out, int out_size, void* d_ws, size_t ws_size,
                              hipStream_t stream) {
  const float* sup    = (const float*)d_in[0];
  const float* qry    = (const float*)d_in[1];
  const int*   labels = (const int*)  d_in[2];
  const float* aw_W1 = (const float*)d_in[3];  const float* aw_b1 = (const float*)d_in[4];
  const float* aw_W2 = (const float*)d_in[5];  const float* aw_b2 = (const float*)d_in[6];
  const float* aw_g  = (const float*)d_in[7];  const float* aw_be = (const float*)d_in[8];
  const float* ct_W1 = (const float*)d_in[9];  const float* ct_b1 = (const float*)d_in[10];
  const float* ct_W2 = (const float*)d_in[11]; const float* ct_b2 = (const float*)d_in[12];
  const float* ct_g  = (const float*)d_in[13]; const float* ct_be = (const float*)d_in[14];
  const float* ad_W1 = (const float*)d_in[15]; const float* ad_b1 = (const float*)d_in[16];
  const float* ad_W2 = (const float*)d_in[17]; const float* ad_b2 = (const float*)d_in[18];
  const float* cl_W1 = (const float*)d_in[19]; const float* cl_b1 = (const float*)d_in[20];
  const float* cl_W2 = (const float*)d_in[21]; const float* cl_b2 = (const float*)d_in[22];
  const float* cl_W3 = (const float*)d_in[23]; const float* cl_b3 = (const float*)d_in[24];

  // ---- workspace carve-out ----
  char* ws = (char*)d_ws;
  size_t off = 0;
  auto carve = [&](size_t bytes) -> char* {
    char* p = ws + off;
    off += (bytes + 255) & ~(size_t)255;
    return p;
  };
  unsigned* w_aw1 = (unsigned*)carve((size_t)HD * HD * 2);      // packed [K/2][N] dwords
  unsigned* w_aw2 = (unsigned*)carve((size_t)HD * HD * 2);
  unsigned* w_ct1 = (unsigned*)carve((size_t)HD * HD * 2);
  unsigned* w_ct2 = (unsigned*)carve((size_t)HD * HD * 2);
  unsigned* w_ad1 = (unsigned*)carve((size_t)HD * HD * 2);
  unsigned* w_ad2 = (unsigned*)carve((size_t)HD * HD * 2);
  unsigned* w_cl1 = (unsigned*)carve((size_t)2 * HD * HD * 2);
  unsigned* w_cl2 = (unsigned*)carve((size_t)HD * 64 * 2);
  bf16_t* bfA   = (bf16_t*)carve((size_t)NTOK * HD * 2);        // bf16 activations in
  bf16_t* bfB   = (bf16_t*)carve((size_t)NTOK * HD * 2);        // bf16 hidden
  bf16_t* comb  = (bf16_t*)carve((size_t)NTOK * 2 * HD * 2);    // [16384, 2048] bf16
  float*  tmp2    = (float*)carve((size_t)NTOK * HD * 4);       // f32 scratch
  float*  adapted = (float*)carve((size_t)NTOK * HD * 4);
  float*  protos  = (float*)carve((size_t)NSEN * HD * 4);
  float*  pn      = (float*)carve((size_t)NSEN * HD * 4);

  auto cvt = [&](const float* in, bf16_t* out, long n) {
    int blocks = (int)((n + 255) / 256); if (blocks > 4096) blocks = 4096;
    k_f2bf<<<blocks, 256, 0, stream>>>(in, out, n);
  };
  auto pack = [&](const float* in, unsigned* out, int K, int N) {
    int total = (K / 2) * N;
    k_f2bf_pack<<<(total + 255) / 256, 256, 0, stream>>>(in, out, K, N);
  };
  auto gemm = [&](const bf16_t* A, const unsigned* Bpk, const float* bias,
                  void* Cout, int M, int N, int K, int act, int outbf) {
    dim3 g(N / BN, M / BM);
    if (act) {
      if (outbf) k_gemm<1, 1><<<g, 256, 0, stream>>>(A, Bpk, bias, Cout, M, N, K);
      else       k_gemm<1, 0><<<g, 256, 0, stream>>>(A, Bpk, bias, Cout, M, N, K);
    } else {
      if (outbf) k_gemm<0, 1><<<g, 256, 0, stream>>>(A, Bpk, bias, Cout, M, N, K);
      else       k_gemm<0, 0><<<g, 256, 0, stream>>>(A, Bpk, bias, Cout, M, N, K);
    }
  };

  // ---- weight pre-pack (fragment-native bf16 pairs) ----
  pack(aw_W1, w_aw1, HD, HD);  pack(aw_W2, w_aw2, HD, HD);
  pack(ct_W1, w_ct1, HD, HD);  pack(ct_W2, w_ct2, HD, HD);
  pack(ad_W1, w_ad1, HD, HD);  pack(ad_W2, w_ad2, HD, HD);
  pack(cl_W1, w_cl1, 2 * HD, HD);
  pack(cl_W2, w_cl2, HD, 64);

  // ---- phase 1: support encoder -> prototypes ----
  cvt(sup, bfA, (long)NSUP * HD);
  gemm(bfA, w_aw1, aw_b1, bfB, NSUP, HD, HD, 1, 1);    // relu(x@W1+b1) -> bf16
  gemm(bfB, w_aw2, aw_b2, tmp2, NSUP, HD, HD, 0, 0);   // @W2+b2 -> f32
  k_ln<0><<<NSUP, 256, 0, stream>>>(tmp2, aw_g, aw_be, tmp2, HD, 0);
  k_proto<<<dim3(NSEN, HD / 256), 256, 0, stream>>>(tmp2, labels, protos);
  k_pnorm<<<NSEN, 256, 0, stream>>>(protos, pn);

  // ---- phase 2: 5-step meta-adaptation ----
  (void)hipMemcpyAsync(adapted, qry, (size_t)NTOK * HD * 4, hipMemcpyDeviceToDevice, stream);
  for (int it = 0; it < 5; ++it) {
    k_attn<<<NTOK, 256, 0, stream>>>(adapted, protos, pn, bfA);   // x' -> bf16
    gemm(bfA, w_ad1, ad_b1, bfB, NTOK, HD, HD, 1, 1);
    gemm(bfB, w_ad2, ad_b2, adapted, NTOK, HD, HD, 0, 0);
  }

  // ---- phase 3: dual encoders + classifier ----
  cvt(adapted, bfA, (long)NTOK * HD);
  gemm(bfA, w_aw1, aw_b1, bfB, NTOK, HD, HD, 1, 1);
  gemm(bfB, w_aw2, aw_b2, tmp2, NTOK, HD, HD, 0, 0);
  k_ln<1><<<NTOK, 256, 0, stream>>>(tmp2, aw_g, aw_be, comb, 2 * HD, 0);
  gemm(bfA, w_ct1, ct_b1, bfB, NTOK, HD, HD, 1, 1);
  gemm(bfB, w_ct2, ct_b2, tmp2, NTOK, HD, HD, 0, 0);
  k_ln<1><<<NTOK, 256, 0, stream>>>(tmp2, ct_g, ct_be, comb, 2 * HD, HD);

  gemm(comb, w_cl1, cl_b1, bfB, NTOK, HD, 2 * HD, 1, 1);   // [16384,2048]@[2048,1024]
  gemm(bfB, w_cl2, cl_b2, tmp2, NTOK, 64, HD, 1, 0);       // -> [16384,64] f32
  k_final<<<(NTOK * 3 + 255) / 256, 256, 0, stream>>>(tmp2, cl_W3, cl_b3, (float*)d_out, NTOK);

  (void)in_sizes; (void)n_in; (void)out_size; (void)ws_size;
}
